// MultiHeadAttention_29188597743836
// MI455X (gfx1250) — matrix operand order repaired and checked by reading, not compiled, not run on hardware
//
#include <hip/hip_runtime.h>
#include <hip/hip_bf16.h>

typedef unsigned short u16;
typedef __attribute__((ext_vector_type(16))) __bf16 v16bf;
typedef __attribute__((ext_vector_type(8)))  float  v8f;
typedef __attribute__((ext_vector_type(16))) unsigned short u16x16;
typedef __attribute__((ext_vector_type(8)))  unsigned short u16x8;

#define Bn 4
#define Hn 16
#define Sn 2048
#define Dn 64
#define En 1024

__device__ __forceinline__ u16 f2bf(float f) {
    unsigned int u = __builtin_bit_cast(unsigned int, f);
    u += 0x7FFFu + ((u >> 16) & 1u);   // round-to-nearest-even
    return (u16)(u >> 16);
}

// Gather 16 bf16 (A/B operand for one lane) as two 16-byte loads.
__device__ __forceinline__ v16bf ld16(const u16* p0, const u16* p1) {
    u16x16 r;
    *reinterpret_cast<u16x8*>(&r)       = *reinterpret_cast<const u16x8*>(p0);
    *(reinterpret_cast<u16x8*>(&r) + 1) = *reinterpret_cast<const u16x8*>(p1);
    return __builtin_bit_cast(v16bf, r);
}

__device__ __forceinline__ v8f wmma_bf16(v16bf a, v16bf b, v8f c) {
    return __builtin_amdgcn_wmma_f32_16x16x32_bf16(false, a, false, b, (short)0, c,
                                                   false, false);
}

// ---------------- prep kernels ----------------
__global__ __launch_bounds__(256) void cvt_bf16_kernel(const float* __restrict__ src,
                                                       u16* __restrict__ dst, int n) {
    int i = blockIdx.x * 256 + threadIdx.x;
    if (i < n) dst[i] = f2bf(src[i]);
}

// W[H,E,D] f32 -> Wt[H,D,E] bf16
__global__ __launch_bounds__(256) void transpose_w_kernel(const float* __restrict__ W,
                                                          u16* __restrict__ Wt) {
    int idx = blockIdx.x * 256 + threadIdx.x;            // [0, H*D*E)
    int h = idx / (Dn * En);
    int rem = idx % (Dn * En);
    int d = rem / En;
    int e = rem % En;
    Wt[idx] = f2bf(W[(size_t)h * En * Dn + (size_t)e * Dn + d]);
}

// ---------------- fused QKV projection ----------------
// grid (B*H, S/128), 256 threads; wave w handles 16 rows. Q scaled by 1/8.
__global__ __launch_bounds__(256) void proj_kernel(const u16* __restrict__ Xb,
                                                   const u16* __restrict__ Wqt,
                                                   const u16* __restrict__ Wkt,
                                                   const u16* __restrict__ Wvt,
                                                   u16* __restrict__ Qb,
                                                   u16* __restrict__ Kb,
                                                   u16* __restrict__ Vtb) {
    int bh = blockIdx.x;
    int b = bh / Hn, h = bh % Hn;
    int wave = threadIdx.x >> 5, lane = threadIdx.x & 31;
    int lo = lane & 15, hi = lane >> 4;
    int s0 = blockIdx.y * 128 + wave * 16;

    const u16* xrow = Xb + ((size_t)b * Sn + (s0 + lo)) * En;
    const u16* wq = Wqt + (size_t)h * Dn * En;
    const u16* wk = Wkt + (size_t)h * Dn * En;
    const u16* wv = Wvt + (size_t)h * Dn * En;

    v8f accQ[4], accK[4], accV[4];
#pragma unroll
    for (int t = 0; t < 4; ++t) { accQ[t] = {}; accK[t] = {}; accV[t] = {}; }

    for (int e0 = 0; e0 < En; e0 += 32) {
        v16bf a = ld16(xrow + e0 + hi * 8, xrow + e0 + 16 + hi * 8);
#pragma unroll
        for (int t = 0; t < 4; ++t) {
            const u16* rq = wq + (size_t)(t * 16 + lo) * En + e0 + hi * 8;
            const u16* rk = wk + (size_t)(t * 16 + lo) * En + e0 + hi * 8;
            const u16* rv = wv + (size_t)(t * 16 + lo) * En + e0 + hi * 8;
            accQ[t] = wmma_bf16(a, ld16(rq, rq + 16), accQ[t]);
            accK[t] = wmma_bf16(a, ld16(rk, rk + 16), accK[t]);
            accV[t] = wmma_bf16(a, ld16(rv, rv + 16), accV[t]);
        }
    }

    size_t base = (size_t)(b * Hn + h) * Sn * Dn;  // also == base of [D,S] layout
#pragma unroll
    for (int t = 0; t < 4; ++t) {
#pragma unroll
        for (int r = 0; r < 8; ++r) {
            int row = s0 + r + 8 * hi;
            int col = t * 16 + lo;
            Qb[base + (size_t)row * Dn + col] = f2bf(accQ[t][r] * 0.125f);
            Kb[base + (size_t)row * Dn + col] = f2bf(accK[t][r]);
            Vtb[base + (size_t)col * Sn + row] = f2bf(accV[t][r]);  // V transposed
        }
    }
}

// ---------------- flash attention ----------------
// grid (B*H, S/128), 256 threads; one wave owns a 16-query block.
__global__ __launch_bounds__(256) void attn_kernel(const u16* __restrict__ Qb,
                                                   const u16* __restrict__ Kb,
                                                   const u16* __restrict__ Vtb,
                                                   u16* __restrict__ Attnb) {
    __shared__ __align__(16) u16 pTile[8][16][32];  // per-wave P staging (C->A layout)

    int bh = blockIdx.x;
    int b = bh / Hn, h = bh % Hn;
    int wave = threadIdx.x >> 5, lane = threadIdx.x & 31;
    int lo = lane & 15, hi = lane >> 4;
    int q0 = blockIdx.y * 128 + wave * 16;

    size_t base = (size_t)(b * Hn + h) * Sn * Dn;
    const u16* qrow = Qb + base + (size_t)(q0 + lo) * Dn;
    v16bf aQ0 = ld16(qrow + hi * 8, qrow + 16 + hi * 8);        // d 0..31
    v16bf aQ1 = ld16(qrow + 32 + hi * 8, qrow + 48 + hi * 8);   // d 32..63

    float m_run[8], l_run[8];
    v8f accO[4];
#pragma unroll
    for (int r = 0; r < 8; ++r) { m_run[r] = -3.0e38f; l_run[r] = 0.0f; }
#pragma unroll
    for (int t = 0; t < 4; ++t) accO[t] = {};

    u16(*pT)[32] = pTile[wave];

    for (int k0 = 0; k0 < q0 + 16; k0 += 32) {
        v8f sc0 = {}, sc1 = {};
        const u16* kr0 = Kb + base + (size_t)(k0 + lo) * Dn;
        const u16* kr1 = Kb + base + (size_t)(k0 + 16 + lo) * Dn;
        sc0 = wmma_bf16(aQ0, ld16(kr0 + hi * 8, kr0 + 16 + hi * 8), sc0);
        sc0 = wmma_bf16(aQ1, ld16(kr0 + 32 + hi * 8, kr0 + 48 + hi * 8), sc0);
        sc1 = wmma_bf16(aQ0, ld16(kr1 + hi * 8, kr1 + 16 + hi * 8), sc1);
        sc1 = wmma_bf16(aQ1, ld16(kr1 + 32 + hi * 8, kr1 + 48 + hi * 8), sc1);

        // causal mask: key > query -> -1e30
#pragma unroll
        for (int r = 0; r < 8; ++r) {
            int q = q0 + r + 8 * hi;
            if (k0 + lo > q)      sc0[r] = -1.0e30f;
            if (k0 + 16 + lo > q) sc1[r] = -1.0e30f;
        }

        // online softmax; row values live across the 16-lane half (xor 1,2,4,8)
        float corr[8];
#pragma unroll
        for (int r = 0; r < 8; ++r) {
            float v = fmaxf(sc0[r], sc1[r]);
            v = fmaxf(v, __shfl_xor(v, 1, 32));
            v = fmaxf(v, __shfl_xor(v, 2, 32));
            v = fmaxf(v, __shfl_xor(v, 4, 32));
            v = fmaxf(v, __shfl_xor(v, 8, 32));
            float mnew = fmaxf(m_run[r], v);
            float p0 = __expf(sc0[r] - mnew);
            float p1 = __expf(sc1[r] - mnew);
            float s = p0 + p1;
            s += __shfl_xor(s, 1, 32);
            s += __shfl_xor(s, 2, 32);
            s += __shfl_xor(s, 4, 32);
            s += __shfl_xor(s, 8, 32);
            corr[r] = __expf(m_run[r] - mnew);
            l_run[r] = l_run[r] * corr[r] + s;
            m_run[r] = mnew;
            pT[r + 8 * hi][lo] = f2bf(p0);
            pT[r + 8 * hi][16 + lo] = f2bf(p1);
        }
#pragma unroll
        for (int t = 0; t < 4; ++t)
#pragma unroll
            for (int r = 0; r < 8; ++r) accO[t][r] *= corr[r];

        // per-wave LDS fence (waves have divergent trip counts -> no __syncthreads)
        asm volatile("s_wait_dscnt 0" ::: "memory");

        const u16* prow = &pT[lo][0];
        v16bf aP = ld16(prow + hi * 8, prow + 16 + hi * 8);
#pragma unroll
        for (int t = 0; t < 4; ++t) {
            const u16* vrow = Vtb + base + (size_t)(t * 16 + lo) * Sn + k0 + hi * 8;
            accO[t] = wmma_bf16(aP, ld16(vrow, vrow + 16), accO[t]);
        }
        asm volatile("s_wait_dscnt 0" ::: "memory");
    }

    float inv[8];
#pragma unroll
    for (int r = 0; r < 8; ++r) inv[r] = 1.0f / l_run[r];
#pragma unroll
    for (int t = 0; t < 4; ++t)
#pragma unroll
        for (int r = 0; r < 8; ++r) {
            int row = q0 + r + 8 * hi;
            Attnb[((size_t)b * Sn + row) * En + h * 64 + t * 16 + lo] =
                f2bf(accO[t][r] * inv[r]);
        }
}

// ---------------- output projection: out = Attn * Wo^T + bo ----------------
// grid (B*S/128, E/64), 256 threads.
__global__ __launch_bounds__(256) void oproj_kernel(const u16* __restrict__ Attnb,
                                                    const u16* __restrict__ Wob,
                                                    const float* __restrict__ bo,
                                                    float* __restrict__ out) {
    int wave = threadIdx.x >> 5, lane = threadIdx.x & 31;
    int lo = lane & 15, hi = lane >> 4;
    int row0 = blockIdx.x * 128 + wave * 16;
    int j0 = blockIdx.y * 64;

    const u16* arow = Attnb + (size_t)(row0 + lo) * En;
    v8f acc[4];
#pragma unroll
    for (int t = 0; t < 4; ++t) acc[t] = {};

    for (int i0 = 0; i0 < En; i0 += 32) {
        v16bf a = ld16(arow + i0 + hi * 8, arow + i0 + 16 + hi * 8);
#pragma unroll
        for (int t = 0; t < 4; ++t) {
            const u16* wrow = Wob + (size_t)(j0 + t * 16 + lo) * En + i0 + hi * 8;
            acc[t] = wmma_bf16(a, ld16(wrow, wrow + 16), acc[t]);
        }
    }
#pragma unroll
    for (int t = 0; t < 4; ++t) {
        float bias = bo[j0 + t * 16 + lo];
#pragma unroll
        for (int r = 0; r < 8; ++r)
            out[(size_t)(row0 + r + 8 * hi) * En + j0 + t * 16 + lo] = acc[t][r] + bias;
    }
}

extern "C" void kernel_launch(void* const* d_in, const int* in_sizes, int n_in,
                              void* d_out, int out_size, void* d_ws, size_t ws_size,
                              hipStream_t stream) {
    (void)in_sizes; (void)n_in; (void)out_size; (void)ws_size;
    const float* X  = (const float*)d_in[0];
    const float* Wq = (const float*)d_in[1];
    const float* Wk = (const float*)d_in[2];
    const float* Wv = (const float*)d_in[3];
    const float* Wo = (const float*)d_in[4];
    const float* bo = (const float*)d_in[5];
    float* out = (float*)d_out;

    char* ws = (char*)d_ws;
    const size_t XB = (size_t)Bn * Sn * En * 2;   // 16 MB
    const size_t WT = (size_t)Hn * Dn * En * 2;   // 2 MB
    const size_t QB = (size_t)Bn * Hn * Sn * Dn * 2;  // 16 MB
    u16* Xb   = (u16*)(ws);
    u16* Wqt  = (u16*)(ws + XB);
    u16* Wkt  = (u16*)(ws + XB + WT);
    u16* Wvt  = (u16*)(ws + XB + 2 * WT);
    u16* Wob  = (u16*)(ws + XB + 3 * WT);
    u16* Qb   = (u16*)(ws + XB + 3 * WT + (size_t)En * En * 2);
    u16* Kb   = (u16*)((char*)Qb + QB);
    u16* Vtb  = (u16*)((char*)Kb + QB);
    u16* Attn = (u16*)((char*)Vtb + QB);

    int nx = Bn * Sn * En;
    cvt_bf16_kernel<<<(nx + 255) / 256, 256, 0, stream>>>(X, Xb, nx);
    int nw = Hn * Dn * En;
    transpose_w_kernel<<<(nw + 255) / 256, 256, 0, stream>>>(Wq, Wqt);
    transpose_w_kernel<<<(nw + 255) / 256, 256, 0, stream>>>(Wk, Wkt);
    transpose_w_kernel<<<(nw + 255) / 256, 256, 0, stream>>>(Wv, Wvt);
    int no = En * En;
    cvt_bf16_kernel<<<(no + 255) / 256, 256, 0, stream>>>(Wo, Wob, no);

    proj_kernel<<<dim3(Bn * Hn, Sn / 128), 256, 0, stream>>>(Xb, Wqt, Wkt, Wvt,
                                                             Qb, Kb, Vtb);
    attn_kernel<<<dim3(Bn * Hn, Sn / 128), 256, 0, stream>>>(Qb, Kb, Vtb, Attn);
    oproj_kernel<<<dim3(Bn * Sn / 128, En / 64), 256, 0, stream>>>(Attn, Wob, bo, out);
}
